// BitAstroGPT_66640712564852
// MI455X (gfx1250) — matrix operand order repaired and checked by reading, not compiled, not run on hardware
//
#include <hip/hip_runtime.h>
#include <hip/hip_bf16.h>
#include <math.h>

// Problem constants (from reference)
constexpr int CV    = 32000;   // vocab
constexpr int CB    = 2;       // batch
constexpr int CT    = 2048;    // seq len
constexpr int CD    = 1024;    // model dim
constexpr int CL    = 4;       // layers
constexpr int CH    = 16;      // heads
constexpr int CHD   = 64;      // head dim
constexpr int CHID  = 2730;    // ffn hidden (int(2*D*4/3))
constexpr int CHIDP = 2752;    // ffn hidden padded to multiple of 32
constexpr int CM    = CB * CT; // 4096 token rows
constexpr float CEPS = 1e-6f;

typedef _Float16 v8h  __attribute__((ext_vector_type(8)));
typedef _Float16 v16h __attribute__((ext_vector_type(16)));
typedef float    v8f  __attribute__((ext_vector_type(8)));
typedef unsigned v4u  __attribute__((ext_vector_type(4)));
typedef unsigned v8u  __attribute__((ext_vector_type(8)));

// ---------------------------------------------------------------------------
// CDNA5 async global->LDS copy (GLOBAL_LOAD_ASYNC_TO_LDS_B128, ASYNCcnt).
// 16 bytes per issuing lane; LDS dest address = low 32 bits of generic ptr.
// ---------------------------------------------------------------------------
__device__ __forceinline__ void async_copy_b128(void* lds_dst, const void* gsrc) {
  asm volatile("global_load_async_to_lds_b128 %0, %1, off"
               :
               : "v"((unsigned)(unsigned long long)lds_dst),
                 "v"((unsigned long long)gsrc)
               : "memory");
}
#define WAIT_ASYNC(n) asm volatile("s_wait_asynccnt %0" :: "i"(n) : "memory")

// ---------------------------------------------------------------------------
// CDNA5 Tensor Data Mover: one TENSOR_LOAD_TO_LDS moves a whole 2D tile
// (tile_dim0=32 f16 elems x tile_dim1=128 rows, row stride = ld elems) from
// global memory into contiguous LDS. D# bitfields per CDNA5 ISA.
// Issued by one wave (EXEC is ignored by TDM); tracked with TENSORcnt.
// ---------------------------------------------------------------------------
__device__ __forceinline__ void tdm_load_2d(void* lds_dst, const void* gsrc,
                                            unsigned stride_elems) {
  unsigned long long ga = (unsigned long long)gsrc;
  v4u g0;
  g0[0] = 1u;                                   // [1:0]=count=1 (valid user D#)
  g0[1] = (unsigned)(unsigned long long)lds_dst;// [63:32] lds_addr (bytes)
  g0[2] = (unsigned)(ga & 0xFFFFFFFFu);         // [95:64] global_addr lo
  g0[3] = (unsigned)((ga >> 32) & 0x01FFFFFFu)  // [120:96] global_addr hi
        | (2u << 30);                           // [127:126] type=2 ("image")
  v8u g1;
  g1[0] = (1u << 16);      // [15:0] wg_mask=0 (not in cluster); [17:16] data_size=1 (2B)
  g1[1] = (32u << 16);     // [47:32] atomic_barrier_addr=0; [63:48] tensor_dim0=32
  g1[2] = (128u << 16);    // [79:64] tensor_dim0_hi=0; [95:80] tensor_dim1=128
  g1[3] = (32u << 16);     // [111:96] tensor_dim1_hi=0; [127:112] tile_dim0=32
  g1[4] = 128u;            // [143:128] tile_dim1=128; [159:144] tile_dim2=0
  g1[5] = stride_elems;    // [191:160] tensor_dim0_stride lo32 (data_size units)
  g1[6] = 0u;              // [207:192] stride hi16; [223:208] tensor_dim1_stride lo
  g1[7] = 0u;              // [255:224] tensor_dim1_stride hi
  asm volatile("tensor_load_to_lds %0, %1" :: "s"(g0), "s"(g1) : "memory");
}
#define WAIT_TENSOR(n) asm volatile("s_wait_tensorcnt %0" :: "i"(n) : "memory")

// ---------------------------------------------------------------------------
// WMMA helpers (CDNA5 v_wmma_f32_16x16x32_f16, wave32)
// ---------------------------------------------------------------------------
__device__ __forceinline__ v8f wmma_f16(v16h a, v16h b, v8f c) {
  // 8 args: (neg_a, A, neg_b, B, c_mod, C, reuse_a, reuse_b)
  return __builtin_amdgcn_wmma_f32_16x16x32_f16(false, a, false, b, (short)0, c,
                                                false, false);
}

// A fragment 16(M)x32(K), f16, tile stored row-major [16][ldk]:
// lane m=lane&15 holds K = kb..kb+7 and kb+16..kb+23, kb = (lane&16)?8:0
__device__ __forceinline__ v16h frag_a(const _Float16* row0, int ldk, int lane) {
  const _Float16* p = row0 + (lane & 15) * ldk + ((lane & 16) ? 8 : 0);
  v8h lo = *(const v8h*)p;
  v8h hi = *(const v8h*)(p + 16);
  return __builtin_shufflevector(lo, hi, 0,1,2,3,4,5,6,7,8,9,10,11,12,13,14,15);
}

// B fragment 32(K)x16(N), f16, from a tile stored [N][ldk] (row n contiguous in k):
// lane n=lane&15 holds K = kb..kb+7 and kb+16..kb+23, kb = (lane&16)?8:0
__device__ __forceinline__ v16h frag_b(const _Float16* col0, int ldk, int lane) {
  const _Float16* p = col0 + (lane & 15) * ldk + ((lane & 16) ? 8 : 0);
  v8h lo = *(const v8h*)p;
  v8h hi = *(const v8h*)(p + 16);
  return __builtin_shufflevector(lo, hi, 0,1,2,3,4,5,6,7,8,9,10,11,12,13,14,15);
}

// ---------------------------------------------------------------------------
// BitNet gamma: deterministic two-pass abs-mean reduction
// ---------------------------------------------------------------------------
__global__ __launch_bounds__(256) void k_absmean_p1(const float* __restrict__ w,
                                                    long n, float* __restrict__ part) {
  __shared__ float red[256];
  float s = 0.f;
  for (long i = (long)blockIdx.x * 256 + threadIdx.x; i < n; i += 256L * 256L)
    s += fabsf(w[i]);
  red[threadIdx.x] = s;
  __syncthreads();
  for (int st = 128; st > 0; st >>= 1) {
    if ((int)threadIdx.x < st) red[threadIdx.x] += red[threadIdx.x + st];
    __syncthreads();
  }
  if (threadIdx.x == 0) part[blockIdx.x] = red[0];
}

__global__ __launch_bounds__(256) void k_absmean_p2(const float* __restrict__ part,
                                                    float inv_n, float* __restrict__ gamma) {
  __shared__ float red[256];
  red[threadIdx.x] = part[threadIdx.x];
  __syncthreads();
  for (int st = 128; st > 0; st >>= 1) {
    if ((int)threadIdx.x < st) red[threadIdx.x] += red[threadIdx.x + st];
    __syncthreads();
  }
  if (threadIdx.x == 0) gamma[0] = fmaxf(red[0] * inv_n, 1e-5f);
}

// Quantize (or plain-convert) fp32 weights to f16 with optional K padding.
__global__ __launch_bounds__(256) void k_quant(const float* __restrict__ w,
                                               const float* __restrict__ gamma,
                                               _Float16* __restrict__ out,
                                               int rows, int cols, int colsp, int do_quant) {
  long i = (long)blockIdx.x * 256 + threadIdx.x;
  long total = (long)rows * colsp;
  if (i >= total) return;
  int r = (int)(i / colsp), c = (int)(i % colsp);
  float v = 0.f;
  if (c < cols) {
    v = w[(long)r * cols + c];
    if (do_quant) {
      float g = gamma[0];
      float q = rintf(v / g);            // round half-to-even like jnp.round
      q = fminf(1.f, fmaxf(-1.f, q));
      v = q * g;                          // ternary {-g,0,g}: exact in f16
    }
  }
  out[i] = (_Float16)v;
}

// ---------------------------------------------------------------------------
// Embedding gather (fp32, one row per block, float4 per thread)
// ---------------------------------------------------------------------------
__global__ __launch_bounds__(256) void k_embed(const float* __restrict__ emb,
                                               const int* __restrict__ idx,
                                               float* __restrict__ x) {
  int row = blockIdx.x, tid = threadIdx.x;
  long src = (long)idx[row] * CD;
  ((float4*)(x + (long)row * CD))[tid] = ((const float4*)(emb + src))[tid];
}

// ---------------------------------------------------------------------------
// RMSNorm fp32 -> f16 (one row per block)
// ---------------------------------------------------------------------------
__global__ __launch_bounds__(256) void k_rmsnorm(const float* __restrict__ x,
                                                 const float* __restrict__ g,
                                                 _Float16* __restrict__ out) {
  __shared__ float red[256];
  int row = blockIdx.x, tid = threadIdx.x;
  const float4 v = ((const float4*)(x + (long)row * CD))[tid];
  red[tid] = v.x * v.x + v.y * v.y + v.z * v.z + v.w * v.w;
  __syncthreads();
  for (int st = 128; st > 0; st >>= 1) {
    if (tid < st) red[tid] += red[tid + st];
    __syncthreads();
  }
  float inv = rsqrtf(red[0] / (float)CD + CEPS);
  const float4 gv = ((const float4*)g)[tid];
  _Float16* o = out + (long)row * CD + tid * 4;
  o[0] = (_Float16)(v.x * inv * gv.x);
  o[1] = (_Float16)(v.y * inv * gv.y);
  o[2] = (_Float16)(v.z * inv * gv.z);
  o[3] = (_Float16)(v.w * inv * gv.w);
}

// ---------------------------------------------------------------------------
// RoPE + layout change: [B,T,H,HD] fp32 -> [B*H, T, HD] f16
// ---------------------------------------------------------------------------
__global__ __launch_bounds__(256) void k_rope(const float* __restrict__ in,
                                              _Float16* __restrict__ out, int apply) {
  long i = (long)blockIdx.x * 256 + threadIdx.x;
  if (i >= (long)CM * CD) return;
  int d = (int)(i & (CHD - 1));
  long rest = i >> 6;               // (b,t,h)
  int h = (int)(rest & (CH - 1));
  long bt = rest >> 4;              // b*T + t
  int t = (int)(bt & (CT - 1));
  int b = (int)(bt >> 11);          // T = 2^11
  float v = in[i];
  if (apply) {
    float fr = (float)t * __powf(10000.f, -((float)(d & 31)) / 32.f);
    float c = __cosf(fr), s = __sinf(fr);
    float other = (d < 32) ? -in[i + 32] : in[i - 32];
    v = v * c + other * s;
  }
  out[((long)(b * CH + h) * CT + t) * CHD + d] = (_Float16)v;
}

// ---------------------------------------------------------------------------
// Generic WMMA GEMM: C[M,N](f32) = A[M,K](f16) * B[N,K]^T(f16) (+ residual)
// 128x128 block tile, K-step 32, 8 waves. Full tiles: depth-2 TDM pipeline —
// wave 0 issues one TENSOR_LOAD_TO_LDS per 128x32 tile (A and B), retired
// with s_wait_tensorcnt; 3 LDS buffers. Ragged edge tiles: single-buffer
// register path with zero fill (no long live ranges, no spills).
// ---------------------------------------------------------------------------
__global__ __launch_bounds__(256) void k_gemm(const _Float16* __restrict__ A, int lda,
                                              const _Float16* __restrict__ B, int ldb,
                                              float* C, int ldc, const float* residual,
                                              int M, int N, int K) {
  __shared__ _Float16 As[3][128][32];
  __shared__ _Float16 Bs[3][128][32];
  const int tid = threadIdx.x, lane = tid & 31, w = tid >> 5;
  const int wm = (w & 3) * 32, wn = (w >> 2) * 64;   // wave tile: 32(M) x 64(N)
  const int bm = blockIdx.y * 128, bn = blockIdx.x * 128;
  const int srow = tid >> 1, scol = (tid & 1) * 16;  // staging map (fallback path)
  const int nk = K >> 5;
  const v8f vz = {0.f,0.f,0.f,0.f,0.f,0.f,0.f,0.f};
  v8f acc[2][4];
  for (int i = 0; i < 2; ++i)
    for (int j = 0; j < 4; ++j) acc[i][j] = vz;

  auto compute = [&](int cur) {
    v16h a0 = frag_a(&As[cur][wm][0], 32, lane);
    v16h a1 = frag_a(&As[cur][wm + 16][0], 32, lane);
#pragma unroll
    for (int nt = 0; nt < 4; ++nt) {
      v16h bf = frag_b(&Bs[cur][wn + nt * 16][0], 32, lane);
      acc[0][nt] = wmma_f16(a0, bf, acc[0][nt]);
      acc[1][nt] = wmma_f16(a1, bf, acc[1][nt]);
    }
  };

  const bool full = (bm + 128 <= M) && (bn + 128 <= N);
  if (full) {
    // ---- depth-2 TDM pipeline: 2 tensor ops per stage, wave 0 only ----
    auto stage = [&](int kt, int buf) {
      tdm_load_2d(&As[buf][0][0], A + (long)bm * lda + kt * 32, (unsigned)lda);
      tdm_load_2d(&Bs[buf][0][0], B + (long)bn * ldb + kt * 32, (unsigned)ldb);
    };
    if (w == 0) {
      stage(0, 0);
      if (nk > 1) stage(1, 1);
    }
    for (int kt = 0; kt < nk; ++kt) {
      // retire stage kt on the issuing wave: keep only newest 2-op stage in flight
      if (w == 0) {
        if (kt + 1 < nk) { WAIT_TENSOR(2); } else { WAIT_TENSOR(0); }
      }
      __syncthreads();   // stage kt LDS data visible; all waves done compute(kt-1)
      if (w == 0 && kt + 2 < nk)
        stage(kt + 2, (kt + 2) % 3);   // overwrites buf (kt-1)%3: safe after barrier
      compute(kt % 3);
    }
  } else {
    // ---- single-buffer register fallback with zero fill (edge tiles only) ----
    const uint4 z4 = make_uint4(0u, 0u, 0u, 0u);
    for (int kt = 0; kt < nk; ++kt) {
      uint4 ra0 = z4, ra1 = z4, rb0 = z4, rb1 = z4;
      int ar = bm + srow;
      if (ar < M) {
        const uint4* p = (const uint4*)(A + (long)ar * lda + kt * 32 + scol);
        ra0 = p[0]; ra1 = p[1];
      }
      int br = bn + srow;
      if (br < N) {
        const uint4* p = (const uint4*)(B + (long)br * ldb + kt * 32 + scol);
        rb0 = p[0]; rb1 = p[1];
      }
      if (kt > 0) __syncthreads();   // everyone done computing previous tile
      *(uint4*)&As[0][srow][scol]       = ra0;
      *(uint4*)(&As[0][srow][scol] + 8) = ra1;
      *(uint4*)&Bs[0][srow][scol]       = rb0;
      *(uint4*)(&Bs[0][srow][scol] + 8) = rb1;
      __syncthreads();
      compute(0);
    }
  }

  // C fragment layout: vgpr i -> row = i + ((lane&16)?8:0); col = lane&15
#pragma unroll
  for (int mt = 0; mt < 2; ++mt)
#pragma unroll
    for (int nt = 0; nt < 4; ++nt) {
      int col = bn + wn + nt * 16 + (lane & 15);
      if (col >= N) continue;
#pragma unroll
      for (int i2 = 0; i2 < 8; ++i2) {
        int row = bm + wm + mt * 16 + i2 + ((lane & 16) ? 8 : 0);
        if (row >= M) continue;
        long o = (long)row * ldc + col;
        float v = acc[mt][nt][i2];
        if (residual) v += residual[o];
        C[o] = v;
      }
    }
}

// ---------------------------------------------------------------------------
// Fused W1/W3 GEMM with SiLU-gate epilogue:
// Out[M,Np](f16) = silu(A*B1^T) * (A*B3^T), zero-padded cols [N,Np)
// Double-buffered per-lane async DMA (6 ops/stage) + prefetch; single-buffer
// register fallback for the ragged edge tile.
// ---------------------------------------------------------------------------
__global__ __launch_bounds__(256) void k_gemm_w13(const _Float16* __restrict__ A,
                                                  const _Float16* __restrict__ B1,
                                                  const _Float16* __restrict__ B3,
                                                  _Float16* __restrict__ Out,
                                                  int M, int N, int Np, int K) {
  __shared__ _Float16 As[2][128][32];
  __shared__ _Float16 B1s[2][128][32];
  __shared__ _Float16 B3s[2][128][32];
  const int tid = threadIdx.x, lane = tid & 31, w = tid >> 5;
  const int wm = (w & 3) * 32, wn = (w >> 2) * 64;
  const int bm = blockIdx.y * 128, bn = blockIdx.x * 128;
  const int srow = tid >> 1, scol = (tid & 1) * 16;
  const int nk = K >> 5;
  const v8f vz = {0.f,0.f,0.f,0.f,0.f,0.f,0.f,0.f};
  v8f acc1[2][4], acc2[2][4];
  for (int i = 0; i < 2; ++i)
    for (int j = 0; j < 4; ++j) { acc1[i][j] = vz; acc2[i][j] = vz; }

  auto compute = [&](int cur) {
    v16h a0 = frag_a(&As[cur][wm][0], 32, lane);
    v16h a1 = frag_a(&As[cur][wm + 16][0], 32, lane);
#pragma unroll
    for (int nt = 0; nt < 4; ++nt) {
      v16h b1f = frag_b(&B1s[cur][wn + nt * 16][0], 32, lane);
      v16h b3f = frag_b(&B3s[cur][wn + nt * 16][0], 32, lane);
      acc1[0][nt] = wmma_f16(a0, b1f, acc1[0][nt]);
      acc1[1][nt] = wmma_f16(a1, b1f, acc1[1][nt]);
      acc2[0][nt] = wmma_f16(a0, b3f, acc2[0][nt]);
      acc2[1][nt] = wmma_f16(a1, b3f, acc2[1][nt]);
    }
  };

  const bool full = (bm + 128 <= M) && (bn + 128 <= N);
  if (full) {
    auto stage = [&](int kt, int buf) {
      const _Float16* ga = A + (long)(bm + srow) * K + kt * 32 + scol;
      async_copy_b128(&As[buf][srow][scol],      ga);
      async_copy_b128(&As[buf][srow][scol + 8],  ga + 8);
      const _Float16* g1 = B1 + (long)(bn + srow) * K + kt * 32 + scol;
      async_copy_b128(&B1s[buf][srow][scol],     g1);
      async_copy_b128(&B1s[buf][srow][scol + 8], g1 + 8);
      const _Float16* g3 = B3 + (long)(bn + srow) * K + kt * 32 + scol;
      async_copy_b128(&B3s[buf][srow][scol],     g3);
      async_copy_b128(&B3s[buf][srow][scol + 8], g3 + 8);
    };
    stage(0, 0);
    WAIT_ASYNC(0);
    __syncthreads();
    int cur = 0;
    for (int kt = 0; kt < nk; ++kt) {
      if (kt + 1 < nk) {
        stage(kt + 1, cur ^ 1);   // DMA overlaps WMMA
        if (kt + 2 < nk) {        // -> global_prefetch_b8
          __builtin_prefetch(A + (long)(bm + srow) * K + (kt + 2) * 32 + scol, 0, 1);
          __builtin_prefetch(B1 + (long)(bn + srow) * K + (kt + 2) * 32 + scol, 0, 1);
        }
      }
      compute(cur);
      if (kt + 1 < nk) {
        WAIT_ASYNC(0);
        __syncthreads();
        cur ^= 1;
      }
    }
  } else {
    const uint4 z4 = make_uint4(0u, 0u, 0u, 0u);
    for (int kt = 0; kt < nk; ++kt) {
      uint4 ra0 = z4, ra1 = z4, rb10 = z4, rb11 = z4, rb30 = z4, rb31 = z4;
      int ar = bm + srow;
      if (ar < M) {
        const uint4* pp = (const uint4*)(A + (long)ar * K + kt * 32 + scol);
        ra0 = pp[0]; ra1 = pp[1];
      }
      int br = bn + srow;
      if (br < N) {
        const uint4* p1 = (const uint4*)(B1 + (long)br * K + kt * 32 + scol);
        rb10 = p1[0]; rb11 = p1[1];
        const uint4* p3 = (const uint4*)(B3 + (long)br * K + kt * 32 + scol);
        rb30 = p3[0]; rb31 = p3[1];
      }
      if (kt > 0) __syncthreads();
      *(uint4*)&As[0][srow][scol]        = ra0;
      *(uint4*)(&As[0][srow][scol] + 8)  = ra1;
      *(uint4*)&B1s[0][srow][scol]       = rb10;
      *(uint4*)(&B1s[0][srow][scol] + 8) = rb11;
      *(uint4*)&B3s[0][srow][scol]       = rb30;
      *(uint4*)(&B3s[0][srow][scol] + 8) = rb31;
      __syncthreads();
      compute(0);
    }
  }

#pragma unroll
  for (int mt = 0; mt < 2; ++mt)
#pragma unroll
    for (int nt = 0; nt < 4; ++nt) {
      int col = bn + wn + nt * 16 + (lane & 15);
      if (col >= Np) continue;
#pragma unroll
      for (int i2 = 0; i2 < 8; ++i2) {
        int row = bm + wm + mt * 16 + i2 + ((lane & 16) ? 8 : 0);
        if (row >= M) continue;
        float av = acc1[mt][nt][i2], bv = acc2[mt][nt][i2];
        float sv = av / (1.f + __expf(-av));         // silu
        float val = (col < N) ? sv * bv : 0.f;       // zero K-padding for W2 GEMM
        Out[(long)row * Np + col] = (_Float16)val;
      }
    }
}

// ---------------------------------------------------------------------------
// Flash attention: one block = one (b,h) x 128 query rows; key blocks of 32.
// K tile staged via async-to-LDS; V staged transposed through registers.
// ---------------------------------------------------------------------------
__global__ __launch_bounds__(256) void k_attn(const _Float16* __restrict__ Q,
                                              const _Float16* __restrict__ Kg,
                                              const _Float16* __restrict__ Vg,
                                              _Float16* __restrict__ Y) {
  __shared__ _Float16 Ks[32][64];    // [key][dim]
  __shared__ _Float16 Vs[64][32];    // transposed: [dim][key]
  __shared__ float    Sld[128][32];  // scores tile
  __shared__ _Float16 Ps[128][32];   // softmax numerators (f16)
  __shared__ float mrow[128], lrow[128], rrow[128];

  const int tid = threadIdx.x, lane = tid & 31, w = tid >> 5;
  const int bh = blockIdx.y;                // 0..B*H-1
  const int b = bh >> 4, h = bh & (CH - 1);
  const int qb = blockIdx.x * 128;
  const long base = (long)bh * CT * CHD;
  const int m = lane & 15;
  const int kb8 = (lane & 16) ? 8 : 0;
  const int rowoff = (lane & 16) ? 8 : 0;

  // Q fragments for this wave's 16 query rows (K = HD = 64 -> 2 frags)
  const _Float16* qrow = Q + base + (long)(qb + w * 16 + m) * CHD;
  v16h aq[2];
#pragma unroll
  for (int kk = 0; kk < 2; ++kk) {
    v8h lo = *(const v8h*)(qrow + kk * 32 + kb8);
    v8h hi = *(const v8h*)(qrow + kk * 32 + kb8 + 16);
    aq[kk] = __builtin_shufflevector(lo, hi, 0,1,2,3,4,5,6,7,8,9,10,11,12,13,14,15);
  }

  const v8f vz = {0.f,0.f,0.f,0.f,0.f,0.f,0.f,0.f};
  v8f o[4] = {vz, vz, vz, vz};
  if (tid < 128) { mrow[tid] = -1e30f; lrow[tid] = 0.f; }
  __syncthreads();

  const int njb = (qb >> 5) + 4;  // causal: keys <= qb+127
  for (int jb = 0; jb < njb; ++jb) {
    const int kbase = jb * 32;
    {  // stage K via async DMA (always a full tile), V transposed via registers
      int r = tid >> 3, seg = (tid & 7) * 8;
      async_copy_b128(&Ks[r][seg], Kg + base + (long)(kbase + r) * CHD + seg);
      const _Float16* pv = Vg + base + (long)(kbase + r) * CHD + seg;
#pragma unroll
      for (int i2 = 0; i2 < 8; ++i2) Vs[seg + i2][r] = pv[i2];
    }
    WAIT_ASYNC(0);
    __syncthreads();

    // S = Q K^T (16x32 per wave -> 2 N-tiles, K-loop over HD)
    v8f s[2];
#pragma unroll
    for (int nt = 0; nt < 2; ++nt) {
      s[nt] = wmma_f16(aq[0], frag_b(&Ks[nt * 16][0], 64, lane), vz);
      s[nt] = wmma_f16(aq[1], frag_b(&Ks[nt * 16][32], 64, lane), s[nt]);
    }
#pragma unroll
    for (int nt = 0; nt < 2; ++nt) {
      int key = kbase + nt * 16 + (lane & 15);
#pragma unroll
      for (int i2 = 0; i2 < 8; ++i2) {
        int qr = qb + w * 16 + i2 + rowoff;
        float sv = (key <= qr) ? s[nt][i2] * 0.125f : -1e30f;  // scale=1/sqrt(64)
        Sld[w * 16 + i2 + rowoff][nt * 16 + (lane & 15)] = sv;
      }
    }
    __syncthreads();

    // online softmax, one thread per query row
    if (tid < 128) {
      float mo = mrow[tid], mx = mo;
#pragma unroll 8
      for (int c = 0; c < 32; ++c) mx = fmaxf(mx, Sld[tid][c]);
      float rr = __expf(mo - mx);
      float sum = 0.f;
#pragma unroll 8
      for (int c = 0; c < 32; ++c) {
        float e = __expf(Sld[tid][c] - mx);
        Ps[tid][c] = (_Float16)e;
        sum += e;
      }
      mrow[tid] = mx;
      lrow[tid] = lrow[tid] * rr + sum;
      rrow[tid] = rr;
    }
    __syncthreads();

    // rescale accumulator, then O += P @ V
#pragma unroll
    for (int i2 = 0; i2 < 8; ++i2) {
      float rr = rrow[w * 16 + i2 + rowoff];
#pragma unroll
      for (int nt = 0; nt < 4; ++nt) o[nt][i2] *= rr;
    }
    v16h ap = frag_a(&Ps[w * 16][0], 32, lane);
#pragma unroll
    for (int nt = 0; nt < 4; ++nt)
      o[nt] = wmma_f16(ap, frag_b(&Vs[nt * 16][0], 32, lane), o[nt]);
    __syncthreads();  // before next key block overwrites Ks/Vs/Ps
  }

  // epilogue: O / l  ->  Y[b, t, h*HD + d]
#pragma unroll
  for (int nt = 0; nt < 4; ++nt) {
    int d = nt * 16 + (lane & 15);
#pragma unroll
    for (int i2 = 0; i2 < 8; ++i2) {
      int qr = qb + w * 16 + i2 + rowoff;
      float l = lrow[w * 16 + i2 + rowoff];
      Y[(long)b * CT * CD + (long)qr * CD + h * CHD + d] = (_Float16)(o[nt][i2] / l);
    }
  }
}

// ---------------------------------------------------------------------------
// Host orchestration
// ---------------------------------------------------------------------------
extern "C" void kernel_launch(void* const* d_in, const int* in_sizes, int n_in,
                              void* d_out, int out_size, void* d_ws, size_t ws_size,
                              hipStream_t stream) {
  (void)in_sizes; (void)n_in; (void)out_size; (void)ws_size;
  const float* emb = (const float*)d_in[0];
  const float* Wq  = (const float*)d_in[1];
  const float* Wk  = (const float*)d_in[2];
  const float* Wv  = (const float*)d_in[3];
  const float* Wo  = (const float*)d_in[4];
  const float* W1  = (const float*)d_in[5];
  const float* W3  = (const float*)d_in[6];
  const float* W2  = (const float*)d_in[7];
  const float* g1  = (const float*)d_in[8];
  const float* g2  = (const float*)d_in[9];
  const float* gf  = (const float*)d_in[10];
  const float* Wlm = (const float*)d_in[11];
  const int*   idx = (const int*)d_in[12];
  float* out = (float*)d_out;

  // carve workspace
  char* p = (char*)d_ws;
  auto alloc = [&](size_t bytes) -> void* {
    void* r = (void*)p;
    p += (bytes + 255) & ~(size_t)255;
    return r;
  };
  float*     gpart = (float*)alloc(256 * sizeof(float));
  float*     gamma = (float*)alloc(256);
  _Float16*  wqh   = (_Float16*)alloc((size_t)CD * CD * 2);
  _Float16*  wkh   = (_Float16*)alloc((size_t)CD * CD * 2);
  _Float16*  wvh   = (_Float16*)alloc((size_t)CD * CD * 2);
  _Float16*  woh   = (_Float16*)alloc((size_t)CD * CD * 2);
  _Float16*  w1h   = (_Float16*)alloc((size_t)CHID * CD * 2);
  _Float16*  w3h   = (_Float16*)alloc((size_t)CHID * CD * 2);
  _Float16*  w2h   = (_Float16*)alloc((size_t)CD * CHIDP * 2);
  _Float16*  wlmh  = (_Float16*)alloc((size_t)CV * CD * 2);
  float*     x     = (float*)alloc((size_t)CM * CD * 4);
  _Float16*  h16   = (_Float16*)alloc((size_t)CM * CD * 2);
  float*     qf    = (float*)alloc((size_t)CM * CD * 4);
  float*     kf    = (float*)alloc((size_t)CM * CD * 4);
  float*     vf    = (float*)alloc((size_t)CM * CD * 4);
  _Float16*  Qr    = (_Float16*)alloc((size_t)CM * CD * 2);
  _Float16*  Kr    = (_Float16*)alloc((size_t)CM * CD * 2);
  _Float16*  Vr    = (_Float16*)alloc((size_t)CM * CD * 2);
  _Float16*  Yh    = (_Float16*)alloc((size_t)CM * CD * 2);
  _Float16*  act   = (_Float16*)alloc((size_t)CM * CHIDP * 2);

  auto quantize = [&](const float* wsrc, int rows, int cols, int colsp,
                      _Float16* dst, int doq) {
    k_absmean_p1<<<256, 256, 0, stream>>>(wsrc, (long)rows * cols, gpart);
    k_absmean_p2<<<1, 256, 0, stream>>>(gpart, 1.0f / ((float)rows * (float)cols), gamma);
    long total = (long)rows * colsp;
    k_quant<<<dim3((unsigned)((total + 255) / 256)), 256, 0, stream>>>(
        wsrc, gamma, dst, rows, cols, colsp, doq);
  };
  auto gemm = [&](const _Float16* A, int lda, const _Float16* B, int ldb,
                  float* C, int ldc, const float* res, int Mm, int Nn, int Kk) {
    dim3 g((unsigned)((Nn + 127) / 128), (unsigned)((Mm + 127) / 128));
    k_gemm<<<g, 256, 0, stream>>>(A, lda, B, ldb, C, ldc, res, Mm, Nn, Kk);
  };

  // x = emb[idx]
  k_embed<<<CM, 256, 0, stream>>>(emb, idx, x);
  // LM head weights: f16 convert once (no bitnet quant on Wlm)
  quantize(Wlm, CV, CD, CD, wlmh, 0);

  const int rblocks = (CM * CD) / 256;
  for (int l = 0; l < CL; ++l) {
    const float* wq_ = Wq + (size_t)l * CD * CD;
    const float* wk_ = Wk + (size_t)l * CD * CD;
    const float* wv_ = Wv + (size_t)l * CD * CD;
    const float* wo_ = Wo + (size_t)l * CD * CD;
    const float* w1_ = W1 + (size_t)l * CHID * CD;
    const float* w3_ = W3 + (size_t)l * CHID * CD;
    const float* w2_ = W2 + (size_t)l * CD * CHID;

    quantize(wq_, CD, CD, CD, wqh, 1);
    quantize(wk_, CD, CD, CD, wkh, 1);
    quantize(wv_, CD, CD, CD, wvh, 1);
    quantize(wo_, CD, CD, CD, woh, 1);
    quantize(w1_, CHID, CD, CD, w1h, 1);
    quantize(w3_, CHID, CD, CD, w3h, 1);
    quantize(w2_, CD, CHID, CHIDP, w2h, 1);  // pad K to 2752, zero-filled

    // attention block
    k_rmsnorm<<<CM, 256, 0, stream>>>(x, g1 + (size_t)l * CD, h16);
    gemm(h16, CD, wqh, CD, qf, CD, nullptr, CM, CD, CD);
    gemm(h16, CD, wkh, CD, kf, CD, nullptr, CM, CD, CD);
    gemm(h16, CD, wvh, CD, vf, CD, nullptr, CM, CD, CD);
    k_rope<<<rblocks, 256, 0, stream>>>(qf, Qr, 1);
    k_rope<<<rblocks, 256, 0, stream>>>(kf, Kr, 1);
    k_rope<<<rblocks, 256, 0, stream>>>(vf, Vr, 0);
    k_attn<<<dim3(CT / 128, CB * CH), 256, 0, stream>>>(Qr, Kr, Vr, Yh);
    gemm(Yh, CD, woh, CD, x, CD, x, CM, CD, CD);  // x = x + attn_out @ Wo^T

    // mlp block
    k_rmsnorm<<<CM, 256, 0, stream>>>(x, g2 + (size_t)l * CD, h16);
    k_gemm_w13<<<dim3((CHID + 127) / 128, CM / 128), 256, 0, stream>>>(
        h16, w1h, w3h, act, CM, CHID, CHIDP, CD);
    gemm(act, CHIDP, w2h, CHIDP, x, CD, x, CM, CD, CHIDP);  // x = x + mlp
  }

  // logits = rmsnorm(x, gf) @ Wlm^T
  k_rmsnorm<<<CM, 256, 0, stream>>>(x, gf, h16);
  gemm(h16, CD, wlmh, CD, out, CV, nullptr, CM, CV, CD);
}
